// ImdbCapsuleNetworkModel_65678639891151
// MI455X (gfx1250) — matrix operand order repaired and checked by reading, not compiled, not run on hardware
//
#include <hip/hip_runtime.h>
#include <hip/hip_bf16.h>

// ---------------------------------------------------------------------------
// MI455X (gfx1250) capsule network forward.
//   conv1/conv2 -> bf16 WMMA GEMMs (v_wmma_f32_16x16x32_bf16, wave32)
//     - weights staged global->LDS with global_load_async_to_lds_b128 (ASYNCcnt)
//     - B fragments software-pipelined against the WMMA chain
//   squash / u_hat / routing -> f32 VALU (memory-bound tail)
// ---------------------------------------------------------------------------

typedef unsigned short u16;
typedef unsigned int   u32;
typedef __attribute__((ext_vector_type(16))) __bf16 bf16x16;
typedef __attribute__((ext_vector_type(8)))  float  f32x8;

struct alignas(16) U4 { u32 x[4]; };

union FragBF { bf16x16 v; U4 q[2]; };

#define CAPS_EPS 1e-8f

// sizes
#define NB      64
#define LSEQ    500
#define EMB     300
#define NF      256
#define CO1     498         // 500-3+1
#define CO2     496         // 498-3+1
#define K1      900         // 3*300
#define K1P     928         // padded to 29*32
#define K2      768         // 3*256 (24*32 exactly)
#define M1      (NB*CO1)    // 31872 = 996*32
#define M2      (NB*CO2)    // 31744 = 992*32
#define INCAPS  15872       // 32*496
#define LDK     40          // padded LDS row (ushorts): 20 dwords, conflict-free

__device__ __forceinline__ u16 f32_to_bf16(float f) {
  u32 u = __float_as_uint(f);
  u32 r = u + 0x7FFFu + ((u >> 16) & 1u);   // round-to-nearest-even
  return (u16)(r >> 16);
}

__device__ __forceinline__ u32 pack2(float a, float b) {
  return (u32)f32_to_bf16(a) | ((u32)f32_to_bf16(b) << 16);
}

// LDS byte offset of a generic pointer to __shared__ (low 32 bits per aperture)
__device__ __forceinline__ u32 lds_off(const void* p) {
  return (u32)(uintptr_t)p;
}

// CDNA5 async global->LDS copy (GVS mode), tracked by ASYNCcnt
__device__ __forceinline__ void async_b128(u32 lds_byte_off, const void* sbase,
                                           u32 gbyte_off) {
  asm volatile("global_load_async_to_lds_b128 %0, %1, %2"
               :
               : "v"(lds_byte_off), "v"(gbyte_off), "s"(sbase)
               : "memory");
}
__device__ __forceinline__ void wait_async0() {
  asm volatile("s_wait_asynccnt 0x0" ::: "memory");
}

// ---------------- weight packing: pre-transposed [n][k] bf16 ----------------
__global__ void k_pack_w1(const float* __restrict__ w, u16* __restrict__ out) {
  int idx = blockIdx.x * 256 + threadIdx.x;      // out[n*928+k]
  int k = idx % K1P;
  int n = idx / K1P;
  float v = (k < K1) ? w[n * K1 + k] : 0.f;      // conv1_w (256,1,3,300): k=fs*300+e
  out[idx] = f32_to_bf16(v);
}

__global__ void k_pack_w2(const float* __restrict__ w, u16* __restrict__ out) {
  int idx = blockIdx.x * 256 + threadIdx.x;      // out[co*768+k], k=dt*256+ci
  int k  = idx % K2;
  int co = idx / K2;
  int dt = k >> 8, ci = k & 255;
  out[idx] = f32_to_bf16(w[co * K2 + ci * 3 + dt]);   // conv2_w (co,ci,3,1)
}

// ---------------- WMMA fragment layouts (CDNA5 ISA 7.12.2) ------------------
// A (16x32 bf16): lane half selects K{+0,+8}; q0 = K[h*8 ..+7], q1 = K[16+h*8..]
// B (32x16 bf16): same K split as A; q0 = K[h*8..+7], q1 = K[16+h*8..]
// C (16x16 f32):  VGPR r -> row r + 8*(lane>>4), col = lane&15

// ---------------- conv1: A=f32 text (implicit im2col), B=W1P, out relu bf16 -
__global__ __launch_bounds__(128) void k_conv1(
    const float* __restrict__ text, const u16* __restrict__ w1p,
    const float* __restrict__ bias, u16* __restrict__ y1) {
  __shared__ __align__(16) u16 lA[32 * LDK];
  __shared__ __align__(16) u16 lB[256 * LDK];

  const int tid   = threadIdx.x;
  const int wave  = tid >> 5, lane = tid & 31;
  const int wm    = wave >> 1, wn = wave & 1;   // wave tile: rows wm*16, cols wn*128
  const int lhalf = lane >> 4, l16 = lane & 15;
  const int blockM = blockIdx.x * 32;

  f32x8 acc[8];
#pragma unroll
  for (int s = 0; s < 8; ++s)
#pragma unroll
    for (int e = 0; e < 8; ++e) acc[s][e] = 0.f;

  // staging assignment
  const int ar = tid >> 2;                 // 0..31 (A row)
  const int ak = (tid & 3) * 8;            // A k-offset 0/8/16/24
  const int am = blockM + ar;
  const int ab = am / CO1, at = am % CO1;
  const float* arow = text + (size_t)ab * (LSEQ * EMB) + (size_t)at * EMB;

  const int bn0 = tid * 2;                 // two weight rows per thread
  const u32 lB_n0 = lds_off(&lB[bn0 * LDK]);
  const u32 lB_n1 = lds_off(&lB[(bn0 + 1) * LDK]);
  const u16* aF0 = &lA[(wm * 16 + l16) * LDK + lhalf * 8];

  auto stage_b = [&](int k0) {
#pragma unroll
    for (int q = 0; q < 4; ++q) {
      async_b128(lB_n0 + q * 16, w1p, (u32)((bn0 * K1P + k0 + q * 8) * 2));
      async_b128(lB_n1 + q * 16, w1p, (u32)(((bn0 + 1) * K1P + k0 + q * 8) * 2));
    }
  };
  auto compute = [&]() {
    FragBF a;
    a.q[0] = *(const U4*)aF0;
    a.q[1] = *(const U4*)(aF0 + 16);
    U4 n0 = *(const U4*)&lB[(wn * 128 + l16) * LDK + lhalf * 8];
    U4 n1 = *(const U4*)&lB[(wn * 128 + l16) * LDK + lhalf * 8 + 16];
#pragma unroll
    for (int s = 0; s < 8; ++s) {
      FragBF cur;
      cur.q[0] = n0;
      cur.q[1] = n1;
      if (s < 7) {  // prefetch next B fragment so the WMMA overlaps ds_loads
        const int n = wn * 128 + (s + 1) * 16 + l16;
        n0 = *(const U4*)&lB[n * LDK + lhalf * 8];
        n1 = *(const U4*)&lB[n * LDK + lhalf * 8 + 16];
      }
      acc[s] = __builtin_amdgcn_wmma_f32_16x16x32_bf16(
          false, a.v, false, cur.v, (short)0, acc[s], false, false);
    }
  };

  // 28 full K-steps: k0+31 <= 895 < 900, fully in-bounds -> vector loads
  for (int kt = 0, k0 = 0; kt < 28; ++kt, k0 += 32) {
    stage_b(k0);
    float4 f0 = *(const float4*)(arow + k0 + ak);
    float4 f1 = *(const float4*)(arow + k0 + ak + 4);
    U4 pk;
    pk.x[0] = pack2(f0.x, f0.y);
    pk.x[1] = pack2(f0.z, f0.w);
    pk.x[2] = pack2(f1.x, f1.y);
    pk.x[3] = pack2(f1.z, f1.w);
    *(U4*)&lA[ar * LDK + ak] = pk;
    __builtin_prefetch(arow + k0 + 32 + ak, 0, 1);  // global_prefetch_b8
    wait_async0();
    __syncthreads();
    compute();
    __syncthreads();
  }
  {  // tail step k0=896: K 896..899 valid, 900..927 zero-padded
    const int k0 = 896;
    stage_b(k0);
    U4 pk;
#pragma unroll
    for (int q = 0; q < 8; q += 2) {
      int gk = k0 + ak + q;
      float f0 = (gk     < K1) ? arow[gk]     : 0.f;
      float f1 = (gk + 1 < K1) ? arow[gk + 1] : 0.f;
      pk.x[q >> 1] = pack2(f0, f1);
    }
    *(U4*)&lA[ar * LDK + ak] = pk;
    wait_async0();
    __syncthreads();
    compute();   // epilogue reads only registers; no trailing barrier needed
  }

  // epilogue: bias + relu -> bf16 Y1[b][t][ch]
#pragma unroll
  for (int s = 0; s < 8; ++s) {
    int n = wn * 128 + s * 16 + l16;
    float bv = bias[n];
#pragma unroll
    for (int r = 0; r < 8; ++r) {
      int m = blockM + wm * 16 + lhalf * 8 + r;
      float v = acc[s][r] + bv;
      v = v > 0.f ? v : 0.f;
      y1[(size_t)m * NF + n] = f32_to_bf16(v);
    }
  }
}

// ---------------- conv2: A=bf16 Y1 (implicit im2col), out f32 caps layout ---
__global__ __launch_bounds__(128) void k_conv2(
    const u16* __restrict__ y1, const u16* __restrict__ w2p,
    const float* __restrict__ bias, float* __restrict__ caps) {
  __shared__ __align__(16) u16 lA[32 * LDK];
  __shared__ __align__(16) u16 lB[256 * LDK];

  const int tid   = threadIdx.x;
  const int wave  = tid >> 5, lane = tid & 31;
  const int wm    = wave >> 1, wn = wave & 1;
  const int lhalf = lane >> 4, l16 = lane & 15;
  const int blockM = blockIdx.x * 32;

  f32x8 acc[8];
#pragma unroll
  for (int s = 0; s < 8; ++s)
#pragma unroll
    for (int e = 0; e < 8; ++e) acc[s][e] = 0.f;

  const int ar = tid >> 2;
  const int ak = (tid & 3) * 8;
  const int am = blockM + ar;
  const int ab = am / CO2, at = am % CO2;
  const u32 a_gbase = (u32)(((size_t)(ab * CO1 + at) * NF) * 2);  // < 16.4MB
  const u32 lA_dst = lds_off(&lA[ar * LDK + ak]);

  const int bn0 = tid * 2;
  const u32 lB_n0 = lds_off(&lB[bn0 * LDK]);
  const u32 lB_n1 = lds_off(&lB[(bn0 + 1) * LDK]);
  const u16* aF0 = &lA[(wm * 16 + l16) * LDK + lhalf * 8];

  for (int kt = 0, k0 = 0; kt < 24; ++kt, k0 += 32) {
    // async stage: B (two weight rows) + A (one bf16 row slice), no VGPR traffic
#pragma unroll
    for (int q = 0; q < 4; ++q) {
      async_b128(lB_n0 + q * 16, w2p, (u32)((bn0 * K2 + k0 + q * 8) * 2));
      async_b128(lB_n1 + q * 16, w2p, (u32)(((bn0 + 1) * K2 + k0 + q * 8) * 2));
    }
    async_b128(lA_dst, y1, a_gbase + (u32)((k0 + ak) * 2));
    wait_async0();
    __syncthreads();

    FragBF a;
    a.q[0] = *(const U4*)aF0;
    a.q[1] = *(const U4*)(aF0 + 16);
    U4 n0 = *(const U4*)&lB[(wn * 128 + l16) * LDK + lhalf * 8];
    U4 n1 = *(const U4*)&lB[(wn * 128 + l16) * LDK + lhalf * 8 + 16];
#pragma unroll
    for (int s = 0; s < 8; ++s) {
      FragBF cur;
      cur.q[0] = n0;
      cur.q[1] = n1;
      if (s < 7) {
        const int n = wn * 128 + (s + 1) * 16 + l16;
        n0 = *(const U4*)&lB[n * LDK + lhalf * 8];
        n1 = *(const U4*)&lB[n * LDK + lhalf * 8 + 16];
      }
      acc[s] = __builtin_amdgcn_wmma_f32_16x16x32_bf16(
          false, a.v, false, cur.v, (short)0, acc[s], false, false);
    }
    __syncthreads();
  }

  // epilogue: +bias, scatter to caps[b][i= (co&31)*496 + t][n= co>>5]
#pragma unroll
  for (int s = 0; s < 8; ++s) {
    int co = wn * 128 + s * 16 + l16;
    float bv = bias[co];
#pragma unroll
    for (int r = 0; r < 8; ++r) {
      int m = blockM + wm * 16 + lhalf * 8 + r;
      int b = m / CO2, t = m % CO2;
      int i = (co & 31) * CO2 + t;
      caps[((size_t)b * INCAPS + i) * 8 + (co >> 5)] = acc[s][r] + bv;
    }
  }
}

// ---------------- squash primary capsules (8-dim) ---------------------------
__global__ void k_squash_caps(float* __restrict__ caps) {
  size_t idx = (size_t)blockIdx.x * 256 + threadIdx.x;   // one capsule / thread
  float* c = caps + idx * 8;
  float s2 = 0.f;
  float v[8];
#pragma unroll
  for (int n = 0; n < 8; ++n) { v[n] = c[n]; s2 += v[n] * v[n]; }
  float sc = s2 / ((1.f + s2) * sqrtf(s2 + CAPS_EPS));
#pragma unroll
  for (int n = 0; n < 8; ++n) c[n] = v[n] * sc;
}

// ---------------- u_hat[b,i,j,m] = sum_n rw[i,j,n,m]*caps[b,i,n] ------------
__global__ __launch_bounds__(256) void k_uhat(
    const float* __restrict__ rw, const float* __restrict__ caps,
    float* __restrict__ uhat) {
  __shared__ float lrw[2048];      // 8 capsules x (2*8*16)
  __shared__ float lcaps[64];      // 8 capsules x 8
  const int i0 = blockIdx.x * 8;
  const float* src = rw + (size_t)i0 * 256;
#pragma unroll
  for (int q = 0; q < 8; ++q)
    lrw[threadIdx.x + q * 256] = src[threadIdx.x + q * 256];

  const int il = threadIdx.x >> 5;
  const int j  = (threadIdx.x >> 4) & 1;
  const int m  = threadIdx.x & 15;
  const float* lw = &lrw[((il * 2 + j) * 8) * 16 + m];

  for (int b = 0; b < NB; ++b) {
    __syncthreads();
    if (threadIdx.x < 64)
      lcaps[threadIdx.x] =
          caps[((size_t)b * INCAPS + i0 + (threadIdx.x >> 3)) * 8 + (threadIdx.x & 7)];
    __syncthreads();
    float acc = 0.f;
#pragma unroll
    for (int n = 0; n < 8; ++n) acc += lw[n * 16] * lcaps[il * 8 + n];
    uhat[((size_t)b * INCAPS + i0 + il) * 32 + j * 16 + m] = acc;
  }
}

// ---------------- routing ---------------------------------------------------
__global__ void k_zero(float* __restrict__ p, int n) {
  int i = blockIdx.x * 256 + threadIdx.x;
  if (i < n) p[i] = 0.f;
}

__global__ __launch_bounds__(256) void k_route_s(
    const float* __restrict__ uhat, const float* __restrict__ blog,
    float* __restrict__ S) {
  const int b = blockIdx.y;
  const int g = threadIdx.x >> 5;
  const int jm = threadIdx.x & 31, j = jm >> 4, m = jm & 15;
  const int i0 = blockIdx.x * 256;
  float acc = 0.f;
  for (int ii = g; ii < 256; ii += 8) {
    size_t bi = (size_t)b * INCAPS + i0 + ii;
    float b0 = blog[bi * 2 + 0], b1 = blog[bi * 2 + 1];
    float mx = fmaxf(b0, b1);
    float e0 = __expf(b0 - mx), e1 = __expf(b1 - mx);
    float c = ((j == 0) ? e0 : e1) / (e0 + e1);
    acc += c * uhat[bi * 32 + j * 16 + m];
  }
  __shared__ float red[8][32];
  red[g][jm] = acc;
  __syncthreads();
  if (threadIdx.x < 32) {
    float t = 0.f;
#pragma unroll
    for (int q = 0; q < 8; ++q) t += red[q][jm];
    atomicAdd(&S[b * 32 + jm], t);     // global_atomic_add_f32
  }
}

__global__ void k_squash_v(const float* __restrict__ S, float* __restrict__ V) {
  int t = threadIdx.x;                // 128 threads: (b,j)
  if (t >= 128) return;
  const float* s = S + t * 16;
  float s2 = 0.f;
#pragma unroll
  for (int m = 0; m < 16; ++m) s2 += s[m] * s[m];
  float sc = s2 / ((1.f + s2) * sqrtf(s2 + CAPS_EPS));
#pragma unroll
  for (int m = 0; m < 16; ++m) V[t * 16 + m] = s[m] * sc;
}

__global__ __launch_bounds__(256) void k_route_b(
    const float* __restrict__ uhat, const float* __restrict__ V,
    float* __restrict__ blog) {
  __shared__ float lv[32];
  const int b = blockIdx.y;
  if (threadIdx.x < 32) lv[threadIdx.x] = V[b * 32 + threadIdx.x];
  __syncthreads();
  size_t bi = (size_t)b * INCAPS + blockIdx.x * 256 + threadIdx.x;
  const float* u = uhat + bi * 32;
  float d0 = 0.f, d1 = 0.f;
#pragma unroll
  for (int m = 0; m < 16; ++m) { d0 += lv[m] * u[m]; d1 += lv[16 + m] * u[16 + m]; }
  blog[bi * 2 + 0] += d0;
  blog[bi * 2 + 1] += d1;
}

__global__ void k_final(const float* __restrict__ V, float* __restrict__ out) {
  int t = blockIdx.x * 256 + threadIdx.x;
  if (t < 2048) out[t] = V[t];
  if (t < NB) {
    float n0 = 0.f, n1 = 0.f;
#pragma unroll
    for (int m = 0; m < 16; ++m) {
      float a = V[t * 32 + m], b = V[t * 32 + 16 + m];
      n0 += a * a; n1 += b * b;
    }
    out[2048 + t] = (n1 > n0) ? 1.f : 0.f;   // argmax (first index on tie)
  }
}

// ---------------------------------------------------------------------------
static inline size_t alignup(size_t x) { return (x + 255) & ~(size_t)255; }

extern "C" void kernel_launch(void* const* d_in, const int* in_sizes, int n_in,
                              void* d_out, int out_size, void* d_ws, size_t ws_size,
                              hipStream_t stream) {
  (void)in_sizes; (void)n_in; (void)out_size; (void)ws_size;
  const float* text = (const float*)d_in[0];
  const float* w1   = (const float*)d_in[1];
  const float* b1   = (const float*)d_in[2];
  const float* w2   = (const float*)d_in[3];
  const float* b2   = (const float*)d_in[4];
  const float* rw   = (const float*)d_in[5];

  char* p = (char*)d_ws;
  u16* W1P = (u16*)p;     p += alignup((size_t)K1P * NF * 2);
  u16* W2P = (u16*)p;     p += alignup((size_t)K2 * NF * 2);
  u16* Y1  = (u16*)p;     p += alignup((size_t)M1 * NF * 2);      // [b][t][ch] bf16
  float* CAPS = (float*)p; p += alignup((size_t)NB * INCAPS * 8 * 4);
  float* UHAT = (float*)p; p += alignup((size_t)NB * INCAPS * 32 * 4);
  float* BLOG = (float*)p; p += alignup((size_t)NB * INCAPS * 2 * 4);
  float* S    = (float*)p; p += alignup(2048 * 4);
  float* V    = (float*)p;

  k_pack_w1<<<(K1P * NF) / 256, 256, 0, stream>>>(w1, W1P);
  k_pack_w2<<<(K2 * NF) / 256, 256, 0, stream>>>(w2, W2P);
  k_conv1<<<M1 / 32, 128, 0, stream>>>(text, W1P, b1, Y1);
  k_conv2<<<M2 / 32, 128, 0, stream>>>(Y1, W2P, b2, CAPS);
  k_squash_caps<<<(NB * INCAPS) / 256, 256, 0, stream>>>(CAPS);
  k_uhat<<<INCAPS / 8, 256, 0, stream>>>(rw, CAPS, UHAT);
  k_zero<<<(NB * INCAPS * 2 + 255) / 256, 256, 0, stream>>>(BLOG, NB * INCAPS * 2);
  for (int it = 0; it < 3; ++it) {
    k_zero<<<8, 256, 0, stream>>>(S, 2048);
    k_route_s<<<dim3(INCAPS / 256, NB), 256, 0, stream>>>(UHAT, BLOG, S);
    k_squash_v<<<1, 128, 0, stream>>>(S, V);
    if (it < 2)
      k_route_b<<<dim3(INCAPS / 256, NB), 256, 0, stream>>>(UHAT, V, BLOG);
  }
  k_final<<<9, 256, 0, stream>>>(V, (float*)d_out);
}
